// Block_12661563588992
// MI455X (gfx1250) — matrix operand order repaired and checked by reading, not compiled, not run on hardware
//
#include <hip/hip_runtime.h>

// ---------------------------------------------------------------------------
// Types matching CDNA5 WMMA / TDM builtin signatures
// ---------------------------------------------------------------------------
typedef __bf16 bf16;
typedef __attribute__((ext_vector_type(16))) __bf16 v16bf;
typedef __attribute__((ext_vector_type(8)))  __bf16 v8bf;
typedef __attribute__((ext_vector_type(8)))  float  v8f;
typedef __attribute__((ext_vector_type(4)))  unsigned int u32x4;
typedef __attribute__((ext_vector_type(8)))  int          i32x8;
typedef __attribute__((ext_vector_type(4)))  int          i32x4;

union V8F   { v8f  v; float f[8]; };
union FragA { v16bf v; v8bf h[2]; };

#define GEMM_PITCH 40   // 32 K elems + 8 pad (16B-aligned rows)
#define ATT_PITCH  72   // 64 elems + 8 pad  (16B-aligned rows)

// Tensor Data Mover availability (device pass only; host pass falls back)
#if defined(__AMDGCN__) && defined(__has_builtin)
#if __has_builtin(__builtin_amdgcn_tensor_load_to_lds) && __has_builtin(__builtin_amdgcn_s_wait_tensorcnt)
#define HAVE_TDM 1
#endif
#endif
#ifndef HAVE_TDM
#define HAVE_TDM 0
#endif

#if HAVE_TDM
// ---------------------------------------------------------------------------
// TDM: 2D tile Global -> LDS with on-the-fly LDS row padding.
// D# packing per CDNA5 ISA (data_size=1 -> 2B).
// ---------------------------------------------------------------------------
__device__ __forceinline__ void tdm_load_2d(unsigned lds_off, const void* gaddr,
                                            unsigned dim0, unsigned dim1,
                                            unsigned tile0, unsigned tile1,
                                            unsigned stride0, unsigned padcfg)
{
    unsigned long long ga = (unsigned long long)(uintptr_t)gaddr;
    u32x4 g0 = { 1u,                                   // count=1 (valid), is_restore=0
                 lds_off,                              // lds_addr [63:32]
                 (unsigned)ga,                         // global_addr [95:64]
                 (unsigned)((ga >> 32) & 0x01FFFFFFu)  // global_addr [120:96]
                   | (2u << 30) };                     // type=2 ("image")
    i32x8 g1 = {
        (int)((1u << 16) | padcfg),                                        // data_size=2B + pad cfg
        (int)((dim0 & 0xFFFFu) << 16),                                     // tensor_dim0 [15:0]
        (int)(((dim0 >> 16) & 0xFFFFu) | ((dim1 & 0xFFFFu) << 16)),        // dim0 hi | dim1 lo
        (int)(((dim1 >> 16) & 0xFFFFu) | ((tile0 & 0xFFFFu) << 16)),       // dim1 hi | tile_dim0
        (int)(tile1 & 0xFFFFu),                                            // tile_dim1 (tile_dim2=0)
        (int)stride0,                                                      // tensor_dim0_stride lo
        0, 0 };
    i32x4 z4 = {0, 0, 0, 0};
#if defined(__clang_major__) && __clang_major__ >= 23
    i32x8 z8 = {0, 0, 0, 0, 0, 0, 0, 0};
    __builtin_amdgcn_tensor_load_to_lds(g0, g1, z4, z4, z8, 0);
#else
    __builtin_amdgcn_tensor_load_to_lds(g0, g1, z4, z4, 0);
#endif
}
// pad config: row = data DWORDs, then pad. interval codes: 3->16 DW, 4->32 DW.
// amount code 3 -> 4 DWORDs (8 bf16) of padding.
#define PADCFG_GEMM ((1u << 20) | (3u << 22) | (3u << 25))   // 32 elem rows -> pitch 40
#define PADCFG_ATT  ((1u << 20) | (4u << 22) | (3u << 25))   // 64 elem rows -> pitch 72
#endif

// ---------------------------------------------------------------------------
// fp32 -> bf16 cast (grid-stride)
// ---------------------------------------------------------------------------
__global__ void cast_f32_bf16(const float* __restrict__ in, bf16* __restrict__ out, int n) {
    int i = blockIdx.x * blockDim.x + threadIdx.x;
    int stride = gridDim.x * blockDim.x;
    for (; i < n; i += stride) out[i] = (bf16)in[i];
}

// ---------------------------------------------------------------------------
// fp32 [R][Cc] -> bf16 transposed [Cc][R]. Block (32,8), LDS-tiled, coalesced.
// ---------------------------------------------------------------------------
__global__ __launch_bounds__(256)
void cast_transpose_f32_bf16(const float* __restrict__ in, bf16* __restrict__ out,
                             int R, int Cc)
{
    __shared__ float tile[32][33];
    const int bx = blockIdx.x * 32;   // column base (of input)
    const int by = blockIdx.y * 32;   // row base (of input)
    const int tx = threadIdx.x, ty = threadIdx.y;
    #pragma unroll
    for (int j = 0; j < 32; j += 8)
        tile[ty + j][tx] = in[(long)(by + ty + j) * Cc + bx + tx];
    __syncthreads();
    #pragma unroll
    for (int j = 0; j < 32; j += 8)
        out[(long)(bx + ty + j) * R + by + tx] = (bf16)tile[tx][ty + j];
}

// ---------------------------------------------------------------------------
// Tiled bf16 GEMM: out = act(A[MxK] @ Wt[NxK]^T + bias)
// Block: 256 thr = 8 waves; block tile 128x128; wave tile 64x32; K-step 32.
// Both tiles staged by TDM (tensor_load_to_lds) into double-buffered LDS;
// wave 0 issues DMA for tile i+1 while all waves run WMMA on tile i.
// ---------------------------------------------------------------------------
__global__ __launch_bounds__(256)
void gemm_bf16_wmma(const bf16* __restrict__ A, const bf16* __restrict__ Wt,
                    const float* __restrict__ bias,
                    float* __restrict__ outF, bf16* __restrict__ outB,
                    int M, int N, int K, int gelu)
{
    __shared__ __align__(16) bf16 As[2][128 * GEMM_PITCH];
    __shared__ __align__(16) bf16 Bs[2][128 * GEMM_PITCH];   // Wt tile: [n][k]

    const int tid    = threadIdx.x;
    const int lane   = tid & 31;
    const int wave   = tid >> 5;
    const int laneLo = lane & 15;
    const int laneHi = lane >> 4;
    const int wm     = wave >> 2;   // 0..1
    const int wn     = wave & 3;    // 0..3
    const long m0    = (long)blockIdx.y * 128;
    const long n0    = (long)blockIdx.x * 128;

    const bf16* Abase = A  + m0 * (long)K;
    const bf16* Bbase = Wt + n0 * (long)K;

    V8F acc[4][2];
    #pragma unroll
    for (int i = 0; i < 4; ++i)
        #pragma unroll
        for (int j = 0; j < 2; ++j)
            #pragma unroll
            for (int r = 0; r < 8; ++r) acc[i][j].f[r] = 0.0f;

#if !HAVE_TDM
    const int srow = tid >> 1;         // 0..127
    const int sseg = (tid & 1) * 16;   // 0 or 16
#endif

    auto stage = [&](int ib, int kt) {
#if HAVE_TDM
        if (wave == 0) {
            tdm_load_2d((unsigned)(uintptr_t)&As[ib][0], Abase + kt,
                        (unsigned)K, (unsigned)M, 32u, 128u, (unsigned)K, PADCFG_GEMM);
            tdm_load_2d((unsigned)(uintptr_t)&Bs[ib][0], Bbase + kt,
                        (unsigned)K, (unsigned)N, 32u, 128u, (unsigned)K, PADCFG_GEMM);
        }
#else
        const bf16* ga = Abase + (long)srow * K + kt + sseg;
        *(v8bf*)&As[ib][srow * GEMM_PITCH + sseg]     = *(const v8bf*)ga;
        *(v8bf*)&As[ib][srow * GEMM_PITCH + sseg + 8] = *(const v8bf*)(ga + 8);
        const bf16* gb = Bbase + (long)srow * K + kt + sseg;
        *(v8bf*)&Bs[ib][srow * GEMM_PITCH + sseg]     = *(const v8bf*)gb;
        *(v8bf*)&Bs[ib][srow * GEMM_PITCH + sseg + 8] = *(const v8bf*)(gb + 8);
#endif
    };

    // prologue: stage first tile, wait, barrier
    stage(0, 0);
#if HAVE_TDM
    if (wave == 0) __builtin_amdgcn_s_wait_tensorcnt(0);
#endif
    __syncthreads();

    int ib = 0;
    for (int kt = 0; kt < K; kt += 32, ib ^= 1) {
        // async-stage next tile into the other buffer (overlaps with WMMA below)
        if (kt + 32 < K) stage(ib ^ 1, kt + 32);

        // --- fragments + WMMA on current buffer ---
        FragA a[4], b[2];
        #pragma unroll
        for (int mt = 0; mt < 4; ++mt) {
            const bf16* p = &As[ib][(wm * 64 + mt * 16 + laneLo) * GEMM_PITCH + laneHi * 8];
            a[mt].h[0] = *(const v8bf*)p;          // K = laneHi*8 .. +7
            a[mt].h[1] = *(const v8bf*)(p + 16);   // K = 16+laneHi*8 .. +7
        }
        #pragma unroll
        for (int nt = 0; nt < 2; ++nt) {
            const bf16* p = &Bs[ib][(wn * 32 + nt * 16 + laneLo) * GEMM_PITCH + laneHi * 8];
            b[nt].h[0] = *(const v8bf*)p;          // K = laneHi*8 .. +7
            b[nt].h[1] = *(const v8bf*)(p + 16);   // K = 16+laneHi*8 .. +7
        }
        #pragma unroll
        for (int mt = 0; mt < 4; ++mt)
            #pragma unroll
            for (int nt = 0; nt < 2; ++nt)
                acc[mt][nt].v = __builtin_amdgcn_wmma_f32_16x16x32_bf16(
                    false, a[mt].v, false, b[nt].v, (short)0, acc[mt][nt].v, false, false);

        // DMA for next tile must be complete before anyone reads it
#if HAVE_TDM
        if (wave == 0) __builtin_amdgcn_s_wait_tensorcnt(0);
#endif
        __syncthreads();
    }

    // --- epilogue: bias, activation, store ---
    float bia[2];
    #pragma unroll
    for (int nt = 0; nt < 2; ++nt)
        bia[nt] = bias ? bias[n0 + wn * 32 + nt * 16 + laneLo] : 0.0f;

    #pragma unroll
    for (int mt = 0; mt < 4; ++mt) {
        #pragma unroll
        for (int nt = 0; nt < 2; ++nt) {
            const long gn = n0 + wn * 32 + nt * 16 + laneLo;
            #pragma unroll
            for (int r = 0; r < 8; ++r) {
                const long gm = m0 + wm * 64 + mt * 16 + r + laneHi * 8;
                float v = acc[mt][nt].f[r] + bia[nt];
                if (gelu) v = 0.5f * v * (1.0f + erff(v * 0.70710678118654752f));
                const long idx = gm * (long)N + gn;
                if (outF) outF[idx] = v;
                if (outB) outB[idx] = (bf16)v;
            }
        }
    }
}

// ---------------------------------------------------------------------------
// Flash attention (causal), bf16 WMMA. Grid: (T/64, B*H). Block: 128 = 4 waves.
// K tile staged via TDM; V tile transposed manually; qkv: [B*T, 3C] bf16.
// ---------------------------------------------------------------------------
__global__ __launch_bounds__(128)
void attn_kernel(const bf16* __restrict__ qkv, bf16* __restrict__ yb)
{
    __shared__ __align__(16) bf16 Qs[64 * ATT_PITCH];
    __shared__ __align__(16) bf16 Ks[64 * ATT_PITCH];
    __shared__ __align__(16) bf16 Vt[64 * ATT_PITCH];   // transposed: [d][k]
    __shared__ __align__(16) bf16 Ps[64 * ATT_PITCH];   // per-wave P staging

    const int tid    = threadIdx.x;
    const int lane   = tid & 31;
    const int wave   = tid >> 5;
    const int laneLo = lane & 15;
    const int laneHi = lane >> 4;
    const int qt = blockIdx.x;          // query tile (64 rows)
    const int bh = blockIdx.y;
    const int b  = bh >> 4;
    const int h  = bh & 15;

    const long rowPitch = 3072;
    const long baseRow  = (long)b * 2048;
    const int  qc = h * 64, kc = 1024 + h * 64, vc = 2048 + h * 64;

    // load Q tile (64x64)
    {
        const int r = tid >> 1, c0 = (tid & 1) * 32;
        const bf16* gp = qkv + (baseRow + qt * 64 + r) * rowPitch + qc + c0;
        #pragma unroll
        for (int j = 0; j < 4; ++j)
            *(v8bf*)&Qs[r * ATT_PITCH + c0 + j * 8] = *(const v8bf*)(gp + j * 8);
    }

    float mrow[8], lrow[8];
    V8F o[4];
    #pragma unroll
    for (int r = 0; r < 8; ++r) { mrow[r] = -__builtin_inff(); lrow[r] = 0.0f; }
    #pragma unroll
    for (int d = 0; d < 4; ++d)
        #pragma unroll
        for (int r = 0; r < 8; ++r) o[d].f[r] = 0.0f;

    for (int kt = 0; kt <= qt; ++kt) {
        __syncthreads();
        // stage K tile (TDM, padded row-major) and V tile (manual transpose)
#if HAVE_TDM
        if (wave == 0) {
            tdm_load_2d((unsigned)(uintptr_t)&Ks[0],
                        qkv + (baseRow + kt * 64) * rowPitch + kc,
                        1u << 20, 1u << 20,
                        64u, 64u, (unsigned)rowPitch, PADCFG_ATT);
        }
#endif
        {
            const int r = tid >> 1, c0 = (tid & 1) * 32;
#if !HAVE_TDM
            const bf16* kp = qkv + (baseRow + kt * 64 + r) * rowPitch + kc + c0;
            #pragma unroll
            for (int j = 0; j < 4; ++j)
                *(v8bf*)&Ks[r * ATT_PITCH + c0 + j * 8] = *(const v8bf*)(kp + j * 8);
#endif
            const bf16* vp = qkv + (baseRow + kt * 64 + r) * rowPitch + vc + c0;
            #pragma unroll
            for (int j = 0; j < 4; ++j) {
                v8bf vv = *(const v8bf*)(vp + j * 8);
                #pragma unroll
                for (int e = 0; e < 8; ++e) Vt[(c0 + j * 8 + e) * ATT_PITCH + r] = vv[e];
            }
        }
#if HAVE_TDM
        if (wave == 0) __builtin_amdgcn_s_wait_tensorcnt(0);
#endif
        __syncthreads();

        // S = Q K^T : 4 tiles of 16 keys, K-reduction over D=64 (2 x K=32)
        FragA qa[2];
        #pragma unroll
        for (int kk = 0; kk < 2; ++kk) {
            const bf16* p = &Qs[(wave * 16 + laneLo) * ATT_PITCH + kk * 32 + laneHi * 8];
            qa[kk].h[0] = *(const v8bf*)p;
            qa[kk].h[1] = *(const v8bf*)(p + 16);
        }
        V8F s[4];
        #pragma unroll
        for (int nt = 0; nt < 4; ++nt) {
            #pragma unroll
            for (int r = 0; r < 8; ++r) s[nt].f[r] = 0.0f;
            #pragma unroll
            for (int kk = 0; kk < 2; ++kk) {
                FragA kb;
                const bf16* p = &Ks[(nt * 16 + laneLo) * ATT_PITCH + kk * 32 + laneHi * 8];
                kb.h[0] = *(const v8bf*)p;
                kb.h[1] = *(const v8bf*)(p + 16);
                s[nt].v = __builtin_amdgcn_wmma_f32_16x16x32_bf16(
                    false, qa[kk].v, false, kb.v, (short)0, s[nt].v, false, false);
            }
        }

        // online softmax (each lane-half owns 8 rows; row spread over 16 lanes)
        const int qbase = qt * 64 + wave * 16 + laneHi * 8;  // + r
        const int kcol  = kt * 64 + laneLo;                  // + nt*16
        #pragma unroll
        for (int r = 0; r < 8; ++r) {
            float v0[4];
            float vmax = -__builtin_inff();
            #pragma unroll
            for (int nt = 0; nt < 4; ++nt) {
                float vv = s[nt].f[r] * 0.125f;              // 1/sqrt(64)
                if (kcol + nt * 16 > qbase + r) vv = -__builtin_inff();
                v0[nt] = vv;
                vmax = fmaxf(vmax, vv);
            }
            #pragma unroll
            for (int m = 1; m < 16; m <<= 1) vmax = fmaxf(vmax, __shfl_xor(vmax, m, 32));
            const float mnew  = fmaxf(mrow[r], vmax);
            const float alpha = __expf(mrow[r] - mnew);
            float rs = 0.0f;
            #pragma unroll
            for (int nt = 0; nt < 4; ++nt) {
                float p = __expf(v0[nt] - mnew);
                rs += p;
                s[nt].f[r] = p;
            }
            #pragma unroll
            for (int m = 1; m < 16; m <<= 1) rs += __shfl_xor(rs, m, 32);
            lrow[r] = lrow[r] * alpha + rs;
            mrow[r] = mnew;
            #pragma unroll
            for (int d = 0; d < 4; ++d) o[d].f[r] *= alpha;
        }

        // stage P (C-layout -> row-major LDS, wave-local rows)
        #pragma unroll
        for (int nt = 0; nt < 4; ++nt)
            #pragma unroll
            for (int r = 0; r < 8; ++r)
                Ps[(wave * 16 + r + laneHi * 8) * ATT_PITCH + nt * 16 + laneLo] = (bf16)s[nt].f[r];

        // O += P @ V
        FragA pa[2];
        #pragma unroll
        for (int kk = 0; kk < 2; ++kk) {
            const bf16* p = &Ps[(wave * 16 + laneLo) * ATT_PITCH + kk * 32 + laneHi * 8];
            pa[kk].h[0] = *(const v8bf*)p;
            pa[kk].h[1] = *(const v8bf*)(p + 16);
        }
        #pragma unroll
        for (int dt = 0; dt < 4; ++dt) {
            #pragma unroll
            for (int kk = 0; kk < 2; ++kk) {
                FragA vb;
                const bf16* p = &Vt[(dt * 16 + laneLo) * ATT_PITCH + kk * 32 + laneHi * 8];
                vb.h[0] = *(const v8bf*)p;
                vb.h[1] = *(const v8bf*)(p + 16);
                o[dt].v = __builtin_amdgcn_wmma_f32_16x16x32_bf16(
                    false, pa[kk].v, false, vb.v, (short)0, o[dt].v, false, false);
            }
        }
    }

    // normalize and store (head-merged [B*T, C] layout)
    #pragma unroll
    for (int dt = 0; dt < 4; ++dt) {
        #pragma unroll
        for (int r = 0; r < 8; ++r) {
            const float val = o[dt].f[r] / lrow[r];
            const long  row = baseRow + qt * 64 + wave * 16 + r + laneHi * 8;
            const int   col = h * 64 + dt * 16 + laneLo;
            yb[row * 1024 + col] = (bf16)val;
        }
    }
}

// ---------------------------------------------------------------------------
// out = res + LayerNorm(a) * g + be ; optional bf16 copy. One block per row.
// ---------------------------------------------------------------------------
__global__ __launch_bounds__(256)
void add_ln_kernel(const float* __restrict__ res, const float* __restrict__ a,
                   const float* __restrict__ g, const float* __restrict__ be,
                   float* __restrict__ outF, bf16* __restrict__ outB, int C)
{
    __shared__ float red[256];
    const long row = blockIdx.x;
    const float* ar = a + row * (long)C;

    float s = 0.0f;
    for (int c = threadIdx.x; c < C; c += 256) s += ar[c];
    red[threadIdx.x] = s; __syncthreads();
    for (int off = 128; off > 0; off >>= 1) {
        if ((int)threadIdx.x < off) red[threadIdx.x] += red[threadIdx.x + off];
        __syncthreads();
    }
    const float mu = red[0] / (float)C;
    __syncthreads();

    float v = 0.0f;
    for (int c = threadIdx.x; c < C; c += 256) { float d = ar[c] - mu; v += d * d; }
    red[threadIdx.x] = v; __syncthreads();
    for (int off = 128; off > 0; off >>= 1) {
        if ((int)threadIdx.x < off) red[threadIdx.x] += red[threadIdx.x + off];
        __syncthreads();
    }
    const float rstd = rsqrtf(red[0] / (float)C + 1e-5f);

    for (int c = threadIdx.x; c < C; c += 256) {
        const float val = res[row * (long)C + c] + (ar[c] - mu) * rstd * g[c] + be[c];
        outF[row * (long)C + c] = val;
        if (outB) outB[row * (long)C + c] = (bf16)val;
    }
}

// ---------------------------------------------------------------------------
// Host launcher
// ---------------------------------------------------------------------------
extern "C" void kernel_launch(void* const* d_in, const int* in_sizes, int n_in,
                              void* d_out, int out_size, void* d_ws, size_t ws_size,
                              hipStream_t stream)
{
    (void)in_sizes; (void)n_in; (void)out_size; (void)ws_size;
    const float* x     = (const float*)d_in[0];
    const float* w_qkv = (const float*)d_in[1];
    const float* b_qkv = (const float*)d_in[2];
    const float* w_out = (const float*)d_in[3];
    const float* b_out = (const float*)d_in[4];
    const float* w1    = (const float*)d_in[5];
    const float* b1    = (const float*)d_in[6];
    const float* w2    = (const float*)d_in[7];
    const float* b2    = (const float*)d_in[8];
    const float* g1    = (const float*)d_in[9];
    const float* be1   = (const float*)d_in[10];
    const float* g2    = (const float*)d_in[11];
    const float* be2   = (const float*)d_in[12];

    const int  B = 4, T = 2048, C = 1024;
    const long BT = (long)B * T;   // 8192

    size_t off = 0;
    auto alloc = [&](size_t bytes) -> void* {
        void* p = (char*)d_ws + off;
        off += (bytes + 255) & ~(size_t)255;
        return p;
    };
    bf16*  xb    = (bf16*) alloc(BT * C * 2);              // x in bf16
    bf16*  wqkvT = (bf16*) alloc((long)3 * C * C * 2);     // [3C][C]
    bf16*  woutT = (bf16*) alloc((long)C * C * 2);         // [C][C]
    bf16*  w1T   = (bf16*) alloc((long)4 * C * C * 2);     // [4C][C]
    bf16*  w2T   = (bf16*) alloc((long)C * 4 * C * 2);     // [C][4C]
    bf16*  qkvB  = (bf16*) alloc(BT * 3 * C * 2);
    bf16*  yB    = (bf16*) alloc(BT * C * 2);
    float* a0    = (float*)alloc(BT * C * 4);
    float* x1    = (float*)alloc(BT * C * 4);
    bf16*  x1B   = (bf16*) alloc(BT * C * 2);
    bf16*  hB    = (bf16*) alloc(BT * 4 * C * 2);
    float* a1    = (float*)alloc(BT * C * 4);

    // fp32 -> bf16 cast (x) and cast+transpose (weights -> [N][K])
    cast_f32_bf16<<<2048, 256, 0, stream>>>(x, xb, (int)(BT * C));
    dim3 tb(32, 8);
    cast_transpose_f32_bf16<<<dim3(3 * C / 32, C / 32),     tb, 0, stream>>>(w_qkv, wqkvT, C, 3 * C);
    cast_transpose_f32_bf16<<<dim3(C / 32,     C / 32),     tb, 0, stream>>>(w_out, woutT, C, C);
    cast_transpose_f32_bf16<<<dim3(4 * C / 32, C / 32),     tb, 0, stream>>>(w1,    w1T,   C, 4 * C);
    cast_transpose_f32_bf16<<<dim3(C / 32,     4 * C / 32), tb, 0, stream>>>(w2,    w2T,   4 * C, C);

    // qkv = x @ w_qkv + b_qkv              [8192, 3072] bf16
    gemm_bf16_wmma<<<dim3(3 * C / 128, BT / 128), 256, 0, stream>>>(
        xb, wqkvT, b_qkv, nullptr, qkvB, (int)BT, 3 * C, C, 0);

    // causal flash attention              y [8192, 1024] bf16
    attn_kernel<<<dim3(T / 64, B * 16), 128, 0, stream>>>(qkvB, yB);

    // a0 = y @ w_out + b_out               fp32
    gemm_bf16_wmma<<<dim3(C / 128, BT / 128), 256, 0, stream>>>(
        yB, woutT, b_out, a0, nullptr, (int)BT, C, C, 0);

    // x1 = x + LN(a0)                      fp32 + bf16
    add_ln_kernel<<<dim3((unsigned)BT), 256, 0, stream>>>(x, a0, g1, be1, x1, x1B, C);

    // h = gelu(x1 @ w1 + b1)               [8192, 4096] bf16
    gemm_bf16_wmma<<<dim3(4 * C / 128, BT / 128), 256, 0, stream>>>(
        x1B, w1T, b1, nullptr, hB, (int)BT, 4 * C, C, 1);

    // a1 = h @ w2 + b2                     fp32
    gemm_bf16_wmma<<<dim3(C / 128, BT / 128), 256, 0, stream>>>(
        hB, w2T, b2, a1, nullptr, (int)BT, C, 4 * C, 0);

    // out = x1 + LN(a1)                    fp32 -> d_out
    add_ln_kernel<<<dim3((unsigned)BT), 256, 0, stream>>>(
        x1, a1, g2, be2, (float*)d_out, nullptr, C);
}
